// MambaKWS_21973052686680
// MI455X (gfx1250) — matrix operand order repaired and checked by reading, not compiled, not run on hardware
//
#include <hip/hip_runtime.h>
#include <hip/hip_bf16.h>
#include <math.h>

#define BSZ 16
#define TSZ 512
#define FEAT_N 40
#define DM 256
#define DI 512
#define DSTATE 16
#define NL 8
#define NC 35
#define NT (BSZ * TSZ)   // 8192 token rows

typedef _Float16 v16h __attribute__((ext_vector_type(16)));
typedef _Float16 h8   __attribute__((ext_vector_type(8)));
typedef float    v8f  __attribute__((ext_vector_type(8)));

// ---------------------------------------------------------------------------
// Generic f32->f16 cast
__global__ void k_cast_f16(const float* __restrict__ src, _Float16* __restrict__ dst, int n) {
    int i = blockIdx.x * blockDim.x + threadIdx.x;
    if (i < n) dst[i] = (_Float16)src[i];
}

// dt_w (NL,512,16) -> f16 padded to (NL,512,32)
__global__ void k_pad_dtw(const float* __restrict__ src, _Float16* __restrict__ dst) {
    int i = blockIdx.x * blockDim.x + threadIdx.x;   // NL*512*32
    if (i >= NL * DI * 32) return;
    int k = i & 31;
    int ld = i >> 5;              // l*512 + d
    dst[i] = (k < 16) ? (_Float16)src[ld * 16 + k] : (_Float16)0.0f;
}

// A = -exp(A_log)
__global__ void k_aneg(const float* __restrict__ src, float* __restrict__ dst, int n) {
    int i = blockIdx.x * blockDim.x + threadIdx.x;
    if (i < n) dst[i] = -__expf(src[i]);
}

// dbl (NT,48) -> dt_lr f16 padded to (NT,32)
__global__ void k_cast_dtlr(const float* __restrict__ dbl, _Float16* __restrict__ dst) {
    int i = blockIdx.x * blockDim.x + threadIdx.x;   // NT*32
    if (i >= NT * 32) return;
    int j = i & 31;
    int row = i >> 5;
    dst[i] = (j < 16) ? (_Float16)dbl[row * 48 + j] : (_Float16)0.0f;
}

// ---------------------------------------------------------------------------
// Input projection + LayerNorm + SiLU  (one block per (b,t) row, 256 threads)
__global__ void k_input_proj(const float* __restrict__ x, const float* __restrict__ pw,
                             const float* __restrict__ pb, const float* __restrict__ g,
                             const float* __restrict__ bt, float* __restrict__ h) {
    int row = blockIdx.x, d = threadIdx.x;
    const float* xr = x + (size_t)row * FEAT_N;
    float acc = pb[d];
    #pragma unroll 8
    for (int f = 0; f < FEAT_N; ++f) acc += xr[f] * pw[d * FEAT_N + f];
    __shared__ float red[DM];
    red[d] = acc; __syncthreads();
    for (int s = 128; s > 0; s >>= 1) { if (d < s) red[d] += red[d + s]; __syncthreads(); }
    float mean = red[0] * (1.0f / DM); __syncthreads();
    float diff = acc - mean;
    red[d] = diff * diff; __syncthreads();
    for (int s = 128; s > 0; s >>= 1) { if (d < s) red[d] += red[d + s]; __syncthreads(); }
    float var = red[0] * (1.0f / DM);
    float y = diff * rsqrtf(var + 1e-5f) * g[d] + bt[d];
    h[(size_t)row * DM + d] = y / (1.0f + __expf(-y));
}

// LayerNorm over DM, optional f16 and/or f32 outputs
__global__ void k_ln_cast(const float* __restrict__ in, const float* __restrict__ g,
                          const float* __restrict__ bt, _Float16* __restrict__ o16,
                          float* __restrict__ o32) {
    int row = blockIdx.x, d = threadIdx.x;
    float v = in[(size_t)row * DM + d];
    __shared__ float red[DM];
    red[d] = v; __syncthreads();
    for (int s = 128; s > 0; s >>= 1) { if (d < s) red[d] += red[d + s]; __syncthreads(); }
    float mean = red[0] * (1.0f / DM); __syncthreads();
    float diff = v - mean;
    red[d] = diff * diff; __syncthreads();
    for (int s = 128; s > 0; s >>= 1) { if (d < s) red[d] += red[d + s]; __syncthreads(); }
    float var = red[0] * (1.0f / DM);
    float y = diff * rsqrtf(var + 1e-5f) * g[d] + bt[d];
    if (o16) o16[(size_t)row * DM + d] = (_Float16)y;
    if (o32) o32[(size_t)row * DM + d] = y;
}

// ---------------------------------------------------------------------------
// Wave-level WMMA GEMM: D[M,N] = A[M,K](f16) * W[N,K](f16)^T (+bias)(+resid)(silu?)
// 128 threads = 4 waves per block, one 16x16 tile per wave. ISA §7.12.2 layouts.
// BOUND=false: all tile loads/stores are in-bounds by construction (no exec churn).
template <bool BOUND>
__global__ void __launch_bounds__(128)
k_wmma_gemm(const _Float16* __restrict__ A, int lda,
            const _Float16* __restrict__ W, int ldb,
            float* __restrict__ D, int ldd,
            const float* __restrict__ resid, int ldr,
            const float* __restrict__ bias,
            _Float16* __restrict__ D16, int ldd16,
            int M, int N, int K, int tilesN, int totalTiles, int act) {
    int tile = blockIdx.x * 4 + (threadIdx.x >> 5);   // wave-uniform
    if (tile >= totalTiles) return;                   // whole-wave exit, EXEC stays full
    int tn = tile % tilesN;
    int tm = tile / tilesN;
    int lane = threadIdx.x & 31;
    int half = lane >> 4;        // 0: lanes 0-15, 1: lanes 16-31
    int r    = lane & 15;

    int arow = tm * 16 + r;      // A rows repeat across halves (M = lane&15)
    int bcol = tn * 16 + r;      // B column (N = lane&15)
    bool aok = !BOUND || (arow < M);
    bool bok = !BOUND || (bcol < N);

    const _Float16* Ap = A + (size_t)arow * lda;
    const _Float16* Wp = W + (size_t)bcol * ldb;

    v8f c = {};
    for (int k0 = 0; k0 < K; k0 += 32) {
        v16h a = {}, b = {};
        if (aok) {
            // elements 0..7  : K = k0 + half*8 + (0..7)
            // elements 8..15 : K = k0 + 16 + half*8 + (0..7)
            h8 a0 = *(const h8*)(Ap + k0 + half * 8);
            h8 a1 = *(const h8*)(Ap + k0 + 16 + half * 8);
            a = __builtin_shufflevector(a0, a1, 0, 1, 2, 3, 4, 5, 6, 7,
                                        8, 9, 10, 11, 12, 13, 14, 15);
            if (k0 + 64 < K) __builtin_prefetch(Ap + k0 + 64, 0, 1);
        }
        if (bok) {
            // elements 0..7 : K = k0 + half*8 + (0..7); elements 8..15 : K = k0 + 16 + half*8 + (0..7)
            h8 b0 = *(const h8*)(Wp + k0 + half * 8);
            h8 b1 = *(const h8*)(Wp + k0 + 16 + half * 8);
            b = __builtin_shufflevector(b0, b1, 0, 1, 2, 3, 4, 5, 6, 7,
                                        8, 9, 10, 11, 12, 13, 14, 15);
        }
        c = __builtin_amdgcn_wmma_f32_16x16x32_f16(false, a, false, b,
                                                   (short)0, c, false, false);
    }

    #pragma unroll
    for (int v = 0; v < 8; ++v) {
        int gm = tm * 16 + v + half * 8;   // C/D: M = v + 8*half
        int gn = tn * 16 + r;              //      N = lane&15
        if (!BOUND || (gm < M && gn < N)) {
            float val = c[v];
            if (bias)  val += bias[gn];
            if (resid) val += resid[(size_t)gm * ldr + gn];
            if (act == 1) val = val / (1.0f + __expf(-val));   // SiLU
            D[(size_t)gm * ldd + gn] = val;
            if (D16) D16[(size_t)gm * ldd16 + gn] = (_Float16)val;
        }
    }
}

// ---------------------------------------------------------------------------
// Causal depthwise conv (k=4) + SiLU on xz[:, :512] -> u16
__global__ void k_conv_silu(const float* __restrict__ xz, const float* __restrict__ cw,
                            const float* __restrict__ cb, _Float16* __restrict__ u16) {
    int idx = blockIdx.x * blockDim.x + threadIdx.x;   // NT*DI
    if (idx >= NT * DI) return;
    int d = idx & (DI - 1);
    int row = idx >> 9;          // b*T + t
    int t = row & (TSZ - 1);
    float acc = cb[d];
    #pragma unroll
    for (int j = 0; j < 4; ++j) {
        int tt = t - 3 + j;
        if (tt >= 0) acc += xz[(size_t)(row - (3 - j)) * 1024 + d] * cw[d * 4 + j];
    }
    u16[idx] = (_Float16)(acc / (1.0f + __expf(-acc)));
}

// ---------------------------------------------------------------------------
// Selective scan: one block per batch element, thread = channel d (512 threads).
// B/C staged through LDS in 128-step chunks. Fuses softplus(dt)+gating.
__global__ void __launch_bounds__(512) k_scan(const _Float16* __restrict__ u16,
                       const float* __restrict__ xz,       // cols 0..511 = dtraw, 512.. = z
                       const float* __restrict__ dbl,      // (NT,48): [16:32)=B, [32:48)=C
                       const float* __restrict__ A,        // (DI,16) = -exp(A_log), layer slice
                       const float* __restrict__ Dp, const float* __restrict__ dtb,
                       _Float16* __restrict__ gated) {
    const int TCH = 128;
    __shared__ float sBC[TCH * 32];
    int b = blockIdx.x, d = threadIdx.x;
    float a[DSTATE], hs[DSTATE];
    #pragma unroll
    for (int s = 0; s < DSTATE; ++s) { a[s] = A[d * DSTATE + s]; hs[s] = 0.0f; }
    float Dpd = Dp[d], dtbd = dtb[d];
    size_t bT = (size_t)b * TSZ;

    for (int c0 = 0; c0 < TSZ; c0 += TCH) {
        __syncthreads();
        for (int i = d; i < TCH * 32; i += DI) {
            int tt = i >> 5, j = i & 31;
            sBC[i] = dbl[(bT + c0 + tt) * 48 + 16 + j];
        }
        __syncthreads();
        for (int tt = 0; tt < TCH; ++tt) {
            size_t row = bT + c0 + tt;
            float u = (float)u16[row * DI + d];
            float draw = xz[row * 1024 + d] + dtbd;
            float dt = (draw > 20.0f) ? draw : log1pf(__expf(draw));
            float du = dt * u;
            float y = 0.0f;
            const float* Bp = &sBC[tt * 32];
            #pragma unroll
            for (int s = 0; s < DSTATE; ++s) {
                hs[s] = hs[s] * __expf(dt * a[s]) + du * Bp[s];
                y += hs[s] * Bp[16 + s];
            }
            y += u * Dpd;
            float z = xz[row * 1024 + DI + d];
            float sz = z / (1.0f + __expf(-z));
            gated[row * DI + d] = (_Float16)(y * sz);
        }
    }
}

// ---------------------------------------------------------------------------
// Masked mean pool over time -> pooled f16 (16 x 256)
__global__ void k_pool(const float* __restrict__ hf, const int* __restrict__ lengths,
                       _Float16* __restrict__ pooled) {
    int b = blockIdx.x, d = threadIdx.x;
    int len = lengths[b]; if (len < 1) len = 1; if (len > TSZ) len = TSZ;
    float s = 0.0f;
    for (int t = 0; t < len; ++t) s += hf[((size_t)b * TSZ + t) * DM + d];
    pooled[(size_t)b * DM + d] = (_Float16)(s / (float)len);
}

// ---------------------------------------------------------------------------
extern "C" void kernel_launch(void* const* d_in, const int* in_sizes, int n_in,
                              void* d_out, int out_size, void* d_ws, size_t ws_size,
                              hipStream_t stream) {
    const float* x        = (const float*)d_in[0];
    const int*   lengths  = (const int*)  d_in[1];
    const float* proj_w   = (const float*)d_in[2];
    const float* proj_b   = (const float*)d_in[3];
    const float* proj_lng = (const float*)d_in[4];
    const float* proj_lnb = (const float*)d_in[5];
    const float* ln_g     = (const float*)d_in[6];
    const float* ln_b     = (const float*)d_in[7];
    const float* in_w     = (const float*)d_in[8];
    const float* conv_w   = (const float*)d_in[9];
    const float* conv_b   = (const float*)d_in[10];
    const float* xp_w     = (const float*)d_in[11];
    const float* dt_w     = (const float*)d_in[12];
    const float* dt_b     = (const float*)d_in[13];
    const float* A_log    = (const float*)d_in[14];
    const float* Dp       = (const float*)d_in[15];
    const float* out_w    = (const float*)d_in[16];
    const float* pre_g    = (const float*)d_in[17];
    const float* pre_b    = (const float*)d_in[18];
    const float* c1_w     = (const float*)d_in[19];
    const float* c1_b     = (const float*)d_in[20];
    const float* c2_w     = (const float*)d_in[21];
    const float* c2_b     = (const float*)d_in[22];
    float* out = (float*)d_out;

    char* ws = (char*)d_ws;
    size_t off = 0;
    auto alloc = [&](size_t bytes) -> void* {
        void* p = ws + off;
        off += (bytes + 255) & ~(size_t)255;
        return p;
    };
    float*     h       = (float*)    alloc((size_t)NT * DM * 4);
    float*     xz      = (float*)    alloc((size_t)NT * 1024 * 4);  // also dtraw + final-LN alias
    _Float16*  xln16   = (_Float16*) alloc((size_t)NT * DM * 2);
    _Float16*  u16     = (_Float16*) alloc((size_t)NT * DI * 2);
    float*     dbl     = (float*)    alloc((size_t)NT * 48 * 4);
    _Float16*  dtlr16  = (_Float16*) alloc((size_t)NT * 32 * 2);
    _Float16*  gated16 = (_Float16*) alloc((size_t)NT * DI * 2);
    _Float16*  inw16   = (_Float16*) alloc((size_t)NL * 1024 * DM * 2);
    _Float16*  xpw16   = (_Float16*) alloc((size_t)NL * 48 * DI * 2);
    _Float16*  dtw16   = (_Float16*) alloc((size_t)NL * DI * 32 * 2);
    _Float16*  outw16  = (_Float16*) alloc((size_t)NL * DM * DI * 2);
    _Float16*  c1w16   = (_Float16*) alloc((size_t)128 * DM * 2);
    _Float16*  c2w16   = (_Float16*) alloc((size_t)NC * 128 * 2);
    float*     Aneg    = (float*)    alloc((size_t)NL * DI * DSTATE * 4);
    _Float16*  pooled  = (_Float16*) alloc((size_t)BSZ * DM * 2);
    float*     z1f     = (float*)    alloc((size_t)BSZ * 128 * 4);
    _Float16*  z116    = (_Float16*) alloc((size_t)BSZ * 128 * 2);
    (void)ws_size; (void)in_sizes; (void)n_in; (void)out_size;

    auto grid1 = [](int n) { return (n + 255) / 256; };
    auto gemmGrid = [](int tiles) { return (tiles + 3) / 4; };   // 4 waves/block

    // Weight preparation (deterministic, every call)
    k_cast_f16<<<grid1(NL*1024*DM), 256, 0, stream>>>(in_w,  inw16,  NL*1024*DM);
    k_cast_f16<<<grid1(NL*48*DI),   256, 0, stream>>>(xp_w,  xpw16,  NL*48*DI);
    k_cast_f16<<<grid1(NL*DM*DI),   256, 0, stream>>>(out_w, outw16, NL*DM*DI);
    k_cast_f16<<<grid1(128*DM),     256, 0, stream>>>(c1_w,  c1w16,  128*DM);
    k_cast_f16<<<grid1(NC*128),     256, 0, stream>>>(c2_w,  c2w16,  NC*128);
    k_pad_dtw <<<grid1(NL*DI*32),   256, 0, stream>>>(dt_w,  dtw16);
    k_aneg    <<<grid1(NL*DI*DSTATE),256,0, stream>>>(A_log, Aneg,   NL*DI*DSTATE);

    // Input projection + LN + SiLU
    k_input_proj<<<NT, DM, 0, stream>>>(x, proj_w, proj_b, proj_lng, proj_lnb, h);

    for (int l = 0; l < NL; ++l) {
        // LN -> f16
        k_ln_cast<<<NT, DM, 0, stream>>>(h, ln_g + l*DM, ln_b + l*DM, xln16, nullptr);
        // in_proj: (8192,256)x(1024,256)^T -> xz (8192,1024)
        {
            int tiles = (NT/16)*(1024/16);
            k_wmma_gemm<false><<<gemmGrid(tiles), 128, 0, stream>>>(
                xln16, DM, inw16 + (size_t)l*1024*DM, DM,
                xz, 1024, nullptr, 0, nullptr, nullptr, 0,
                NT, 1024, DM, 1024/16, tiles, 0);
        }
        // causal conv + SiLU -> u16
        k_conv_silu<<<grid1(NT*DI), 256, 0, stream>>>(xz, conv_w + (size_t)l*DI*4,
                                                      conv_b + (size_t)l*DI, u16);
        // x_proj: (8192,512)x(48,512)^T -> dbl (8192,48)
        {
            int tiles = (NT/16)*3;
            k_wmma_gemm<false><<<gemmGrid(tiles), 128, 0, stream>>>(
                u16, DI, xpw16 + (size_t)l*48*DI, DI,
                dbl, 48, nullptr, 0, nullptr, nullptr, 0,
                NT, 48, DI, 3, tiles, 0);
        }
        // dt_lr -> f16 padded K=32
        k_cast_dtlr<<<grid1(NT*32), 256, 0, stream>>>(dbl, dtlr16);
        // dt_proj: (8192,32)x(512,32)^T -> dtraw, aliased into xz[:, :512] (ldd=1024)
        {
            int tiles = (NT/16)*(DI/16);
            k_wmma_gemm<false><<<gemmGrid(tiles), 128, 0, stream>>>(
                dtlr16, 32, dtw16 + (size_t)l*DI*32, 32,
                xz, 1024, nullptr, 0, nullptr, nullptr, 0,
                NT, DI, 32, DI/16, tiles, 0);
        }
        // selective scan + softplus + SiLU gating -> gated16
        k_scan<<<BSZ, DI, 0, stream>>>(u16, xz, dbl,
                                       Aneg + (size_t)l*DI*DSTATE,
                                       Dp + (size_t)l*DI, dt_b + (size_t)l*DI, gated16);
        // out_proj + residual: h += gated @ out_w^T
        {
            int tiles = (NT/16)*(DM/16);
            k_wmma_gemm<false><<<gemmGrid(tiles), 128, 0, stream>>>(
                gated16, DI, outw16 + (size_t)l*DM*DI, DI,
                h, DM, h, DM, nullptr, nullptr, 0,
                NT, DM, DI, DM/16, tiles, 0);
        }
    }

    // Final LN (f32 into xz alias), masked mean pool
    k_ln_cast<<<NT, DM, 0, stream>>>(h, pre_g, pre_b, nullptr, xz);
    k_pool<<<BSZ, DM, 0, stream>>>(xz, lengths, pooled);

    // Classifier: c1 (16,256)->(16,128) SiLU, then c2 -> (16,35)
    k_wmma_gemm<false><<<gemmGrid(8), 128, 0, stream>>>(
        pooled, DM, c1w16, DM,
        z1f, 128, nullptr, 0, c1_b, z116, 128,
        BSZ, 128, DM, 128/16, 8, 1);
    k_wmma_gemm<true><<<gemmGrid(3), 128, 0, stream>>>(
        z116, 128, c2w16, 128,
        out, NC, nullptr, 0, c2_b, nullptr, 0,
        BSZ, NC, 128, 3, 3, 0);
}
